// SchNet_24970939859125
// MI455X (gfx1250) — matrix operand order repaired and checked by reading, not compiled, not run on hardware
//
#include <hip/hip_runtime.h>
#include <hip/hip_bf16.h>
#include <math.h>

typedef __attribute__((ext_vector_type(16))) _Float16 v16h;
typedef __attribute__((ext_vector_type(8)))  float    v8f;

#define Bq 16
#define Aq 128
#define Nq 127
#define Fq 128
#define ATOMS (Bq*Aq)          // 2048
#define LOG2F_ 0.6931471805599453f
#define CUT 5.0f

__device__ __forceinline__ float sspf(float x) {
    // shifted softplus via hardware v_exp_f32 / v_log_f32:
    // softplus(x) = max(x,0) + log(1 + exp(-|x|));   ssp = softplus - log2
    return fmaxf(x, 0.0f) + __logf(1.0f + __expf(-fabsf(x))) - LOG2F_;
}

__device__ __forceinline__ v8f wmma16(v16h a, v16h b, v8f c) {
    return __builtin_amdgcn_wmma_f32_16x16x32_f16(
        false, a, false, b, (short)0, c, false, false);
}

// ---------------------------------------------------------------------------
// Generic f32 -> f16 convert (row-major weights for B-fragments)
// ---------------------------------------------------------------------------
__global__ void schnet_cvt_k(const float* __restrict__ src,
                             _Float16* __restrict__ dst, int n) {
    int idx = blockIdx.x * blockDim.x + threadIdx.x;
    if (idx < n) dst[idx] = (_Float16)src[idx];
}

// W1 [3][25][128] -> f16 padded to [3][32][128] (zero rows 25..31)
__global__ void schnet_cvtW1_k(const float* __restrict__ W1,
                               _Float16* __restrict__ W1h) {
    int idx = blockIdx.x * blockDim.x + threadIdx.x;
    if (idx >= 3 * 32 * 128) return;
    int l = idx / (32 * 128);
    int rem = idx - l * (32 * 128);
    int k = rem >> 7, c = rem & 127;
    W1h[idx] = (_Float16)((k < 25) ? W1[(l * 25 + k) * 128 + c] : 0.0f);
}

// ---------------------------------------------------------------------------
// Repack a [2048,128] f32 activation into f16 WMMA A-fragment order:
// frag[((mt*4+kt)*32 + lane)*16 + i], lane = M + 16*((K2%16)>=8),
// i = (K2%8) + 8*(K2/16)
// ---------------------------------------------------------------------------
__global__ void schnet_x2frag_k(const float* __restrict__ X,
                                _Float16* __restrict__ Xh) {
    int idx = blockIdx.x * blockDim.x + threadIdx.x;
    if (idx >= ATOMS * 128) return;
    int row = idx >> 7, k = idx & 127;
    int mt = row >> 4, M2 = row & 15;
    int kt = k >> 5, K2 = k & 31;
    int lane = M2 + ((((K2 & 15) >= 8)) ? 16 : 0);
    int i = (K2 & 7) + ((K2 >> 4) << 3);
    Xh[(((size_t)(mt * 4 + kt) * 32 + lane) << 4) + i] = (_Float16)X[idx];
}

// ---------------------------------------------------------------------------
// WMMA GEMM for the small M=2048, K=N=128 layers.
// grid = 128 row-tiles; 8 waves/block, wave w -> 16-col strip; 4 WMMA (K=128).
// out = act(Xh @ Wh + bias) (+ resid), fp32 out.
// ---------------------------------------------------------------------------
__global__ void __launch_bounds__(256)
schnet_gemm_wmma_k(const _Float16* __restrict__ Xh,   // A-fragment order
                   const _Float16* __restrict__ Wh,   // [128][128] row-major
                   const float*    __restrict__ bias, // nullable
                   const float*    __restrict__ resid,// nullable
                   float*          __restrict__ out,
                   int act)
{
    const int mt   = blockIdx.x;
    const int tid  = threadIdx.x;
    const int lane = tid & 31;
    const int wave = tid >> 5;
    const int c0   = wave << 4;
    const int nsub = lane & 15;
    const int mhi  = (lane >> 4) << 3;

    v16h bf[4];
#pragma unroll
    for (int kt = 0; kt < 4; ++kt)
        for (int s = 0; s < 16; ++s)
            bf[kt][s] = Wh[(size_t)(kt * 32 + ((s & 8) << 1) + ((lane >> 4) << 3) + (s & 7)) * 128 + c0 + nsub];

    v8f acc = {};
#pragma unroll
    for (int kt = 0; kt < 4; ++kt) {
        v16h a = *(const v16h*)(Xh + (((size_t)(mt * 4 + kt) * 32 + lane) << 4));
        acc = wmma16(a, bf[kt], acc);
    }

    float bv = bias ? bias[c0 + nsub] : 0.0f;
#pragma unroll
    for (int r = 0; r < 8; ++r) {
        int row = mt * 16 + r + mhi;
        float v = acc[r] + bv;
        if (act) v = sspf(v);
        size_t o = ((size_t)row << 7) + c0 + nsub;
        if (resid) v += resid[o];
        out[o] = v;
    }
}

// ---------------------------------------------------------------------------
// Distances + gaussian smearing (staged as WMMA A-fragments, f16, K padded
// to 32) + cosine-cutoff*mask (Cm).  grid = ATOMS, block = 128 (n index).
// ---------------------------------------------------------------------------
__global__ void schnet_prep_k(const float* __restrict__ pos,
                              const float* __restrict__ cell,
                              const float* __restrict__ coff,
                              const float* __restrict__ mask,
                              const int*   __restrict__ nbrs,
                              _Float16* __restrict__ Fg,   // [ATOMS][4096] frag order
                              float*    __restrict__ CmG)  // [ATOMS][128]
{
    int atom = blockIdx.x;             // b*128 + a
    int b    = atom >> 7;
    int n    = threadIdx.x;            // 0..127, row 127 = pad
    float r = 0.0f, Cmv = 0.0f;
    bool valid = (n < Nq);
    if (valid) {
        int nb = nbrs[(size_t)atom * Nq + n];
        float px = pos[atom*3+0], py = pos[atom*3+1], pz = pos[atom*3+2];
        int qi = (b * Aq + nb) * 3;
        float dx = pos[qi+0] - px, dy = pos[qi+1] - py, dz = pos[qi+2] - pz;
        size_t ci = ((size_t)atom * Nq + n) * 3;
        float o0 = coff[ci+0], o1 = coff[ci+1], o2 = coff[ci+2];
        const float* cb = cell + b * 9;
        dx += o0*cb[0] + o1*cb[3] + o2*cb[6];
        dy += o0*cb[1] + o1*cb[4] + o2*cb[7];
        dz += o0*cb[2] + o1*cb[5] + o2*cb[8];
        float d2 = dx*dx + dy*dy + dz*dz;
        float mk = mask[(size_t)atom * Nq + n];
        r = (mk > 0.0f) ? sqrtf(d2) : 0.0f;
        float C = 0.5f * (cosf(r * (float)M_PI / CUT) + 1.0f);
        Cmv = ((r < CUT) ? C : 0.0f) * mk;
    }
    CmG[atom * 128 + n] = Cmv;

    // gaussians, written in 16x32 f16 A-fragment layout
    const float w = CUT / 24.0f;
    const float coeff = -0.5f / (w * w);
    size_t base = (size_t)atom * 4096;
    int m = n >> 4, M = n & 15;
#pragma unroll
    for (int g = 0; g < 32; ++g) {
        float val = 0.0f;
        if (valid && g < 25) {
            float diff = r - (float)g * w;
            val = __expf(coeff * diff * diff);
        }
        int lane = M + (((g & 15) >= 8) ? 16 : 0);
        int i    = (g & 7) + ((g >> 4) << 3);
        Fg[base + (((size_t)(m * 32 + lane)) << 4) + i] = (_Float16)val;
    }
}

// ---------------------------------------------------------------------------
// x = embeddings[atomic_numbers]
// ---------------------------------------------------------------------------
__global__ void schnet_embed_k(const float* __restrict__ emb,
                               const int* __restrict__ zn,
                               float* __restrict__ x) {
    int idx = blockIdx.x * blockDim.x + threadIdx.x;
    if (idx >= ATOMS * Fq) return;
    int af = idx >> 7, f = idx & 127;
    x[idx] = emb[zn[af] * Fq + f];
}

// ---------------------------------------------------------------------------
// Fused filter network + cfconv aggregation, one workgroup per (b,a) atom.
// 8 waves; wave w owns output column strip [16w, 16w+16).
//   Stage 1 (WMMA K=32): H = ssp(Fg @ W1h + b1)   -> LDS in A-frag layout
//   Stage 2 (4x WMMA K=32): Wt = (H @ W2h + b2) * Cm
//   Stage 3: agg[f] = sum_n y[nbr[n], f] * Wt[n, f]   (gather fused, no HBM H)
// ---------------------------------------------------------------------------
__global__ void __launch_bounds__(256)
schnet_filter_agg_k(const _Float16* __restrict__ Fg,     // [ATOMS][4096]
                    const float*    __restrict__ CmG,    // [ATOMS][128]
                    const float*    __restrict__ Y,      // [ATOMS][128]
                    const int*      __restrict__ nbrs,   // [ATOMS][127]
                    const _Float16* __restrict__ W1h,    // [32][128] (layer)
                    const float*    __restrict__ b1,     // [128]
                    const _Float16* __restrict__ W2h,    // [128][128] (layer)
                    const float*    __restrict__ b2,     // [128]
                    float*          __restrict__ aggG)   // [ATOMS][128]
{
    __shared__ __align__(32) _Float16 sH[8 * 4 * 32 * 16]; // 32 KB H frags
    __shared__ float sCm[128];
    __shared__ int   sNbr[128];

    const int atom = blockIdx.x;
    const int bb   = atom >> 7;
    const int tid  = threadIdx.x;
    const int lane = tid & 31;
    const int wave = tid >> 5;
    const int c0   = wave << 4;          // column strip base
    const int nsub = lane & 15;          // N within tile
    const int mhi  = (lane >> 4) << 3;   // +8 rows for upper half-wave

    if (tid < 128) {
        sCm[tid]  = CmG[(size_t)atom * 128 + tid];
        sNbr[tid] = (tid < Nq) ? nbrs[(size_t)atom * Nq + tid] : 0;
    }

    // ---- Stage 1: H = ssp(Fg @ W1 + b1), write to LDS in A-frag layout ----
    v16h bf1;  // lane%16 = N col, elements over K (table order)
    for (int s = 0; s < 16; ++s)
        bf1[s] = W1h[(size_t)(((s & 8) << 1) + ((lane >> 4) << 3) + (s & 7)) * 128 + c0 + nsub];
    float b1v = b1[c0 + nsub];
    const int kt_out = wave >> 1;
    const int i2base = (wave & 1) << 3;
    const int l2add  = (nsub >= 8) ? 16 : 0;
#pragma unroll
    for (int m = 0; m < 8; ++m) {
        v16h a = *(const v16h*)(Fg + (size_t)atom * 4096 +
                                (((size_t)(m * 32 + lane)) << 4));
        v8f acc = {};
        acc = wmma16(a, bf1, acc);
#pragma unroll
        for (int r = 0; r < 8; ++r) {
            float h = sspf(acc[r] + b1v);
            int M2    = r + mhi;                 // row within tile
            int lane2 = M2 + l2add;
            int i2    = (nsub & 7) + i2base;
            sH[((m * 4 + kt_out) * 32 + lane2) * 16 + i2] = (_Float16)h;
        }
    }
    __syncthreads();

    // ---- Stage 2+3: Wt = (H @ W2 + b2) * Cm ; agg += y_nbh * Wt ----
    v16h bf2[4];
#pragma unroll
    for (int kt = 0; kt < 4; ++kt)
        for (int s = 0; s < 16; ++s)
            bf2[kt][s] = W2h[(size_t)(kt * 32 + ((s & 8) << 1) + ((lane >> 4) << 3) + (s & 7)) * 128 + c0 + nsub];
    float b2v = b2[c0 + nsub];
    float aggp = 0.0f;
#pragma unroll
    for (int m = 0; m < 8; ++m) {
        v8f acc = {};
#pragma unroll
        for (int kt = 0; kt < 4; ++kt) {
            v16h a = *(const v16h*)&sH[((m * 4 + kt) * 32 + lane) * 16];
            acc = wmma16(a, bf2[kt], acc);
        }
#pragma unroll
        for (int r = 0; r < 8; ++r) {
            int row = m * 16 + r + mhi;                  // neighbor index n
            float wt = (acc[r] + b2v) * sCm[row];        // pad row -> Cm=0
            int nb = sNbr[row];
            float yv = Y[((size_t)(bb * Aq + nb) << 7) + c0 + nsub];
            aggp += yv * wt;
        }
    }
    // lanes L and L^16 cover complementary row halves of every tile
    aggp += __shfl_xor(aggp, 16);
    if (lane < 16)
        aggG[(size_t)atom * 128 + c0 + lane] = aggp;
}

// ---------------------------------------------------------------------------
extern "C" void kernel_launch(void* const* d_in, const int* in_sizes, int n_in,
                              void* d_out, int out_size, void* d_ws, size_t ws_size,
                              hipStream_t stream) {
    const float* positions = (const float*)d_in[0];
    const float* cell      = (const float*)d_in[1];
    const float* coff      = (const float*)d_in[2];
    const float* nmask     = (const float*)d_in[3];
    const float* emb       = (const float*)d_in[4];
    const float* filt_W1   = (const float*)d_in[5];
    const float* filt_b1   = (const float*)d_in[6];
    const float* filt_W2   = (const float*)d_in[7];
    const float* filt_b2   = (const float*)d_in[8];
    const float* in2f_W    = (const float*)d_in[9];
    const float* f2out_W   = (const float*)d_in[10];
    const float* f2out_b   = (const float*)d_in[11];
    const float* dense_W   = (const float*)d_in[12];
    const float* dense_b   = (const float*)d_in[13];
    const int*   atomnum   = (const int*)d_in[14];
    const int*   neighbors = (const int*)d_in[15];
    float* x = (float*)d_out;               // [B,A,F], running features

    // ---- carve workspace (aligned to 256B) ----
    char* ws = (char*)d_ws;
    size_t off = 0;
    auto carve = [&](size_t bytes) -> char* {
        char* p = ws + off;
        off += (bytes + 255) & ~(size_t)255;
        return p;
    };
    _Float16* Fg   = (_Float16*)carve((size_t)ATOMS * 4096 * sizeof(_Float16)); // 16.8MB
    float*    Cm   = (float*)   carve((size_t)ATOMS * 128 * sizeof(float));
    float*    Y    = (float*)   carve((size_t)ATOMS * 128 * sizeof(float));
    float*    AGG  = (float*)   carve((size_t)ATOMS * 128 * sizeof(float));
    float*    TMP  = (float*)   carve((size_t)ATOMS * 128 * sizeof(float));
    _Float16* Xh   = (_Float16*)carve((size_t)ATOMS * 128 * sizeof(_Float16)); // frag scratch
    _Float16* W1h  = (_Float16*)carve((size_t)3 * 32 * 128 * sizeof(_Float16));
    _Float16* W2h  = (_Float16*)carve((size_t)3 * 128 * 128 * sizeof(_Float16));
    _Float16* I2Fh = (_Float16*)carve((size_t)3 * 128 * 128 * sizeof(_Float16));
    _Float16* F2Oh = (_Float16*)carve((size_t)3 * 128 * 128 * sizeof(_Float16));
    _Float16* DENh = (_Float16*)carve((size_t)3 * 128 * 128 * sizeof(_Float16));
    (void)ws_size; (void)in_sizes; (void)n_in; (void)out_size;

    // ---- weight conversions (all 3 layers, tiny) ----
    const int wN = 3 * 128 * 128;
    schnet_cvtW1_k<<<(3*32*128 + 255)/256, 256, 0, stream>>>(filt_W1, W1h);
    schnet_cvt_k<<<(wN + 255)/256, 256, 0, stream>>>(filt_W2, W2h, wN);
    schnet_cvt_k<<<(wN + 255)/256, 256, 0, stream>>>(in2f_W, I2Fh, wN);
    schnet_cvt_k<<<(wN + 255)/256, 256, 0, stream>>>(f2out_W, F2Oh, wN);
    schnet_cvt_k<<<(wN + 255)/256, 256, 0, stream>>>(dense_W, DENh, wN);

    // distances + gaussian fragments + cutoff*mask
    schnet_prep_k<<<ATOMS, 128, 0, stream>>>(
        positions, cell, coff, nmask, neighbors, Fg, Cm);
    // embedding gather -> x
    schnet_embed_k<<<(ATOMS * Fq + 255) / 256, 256, 0, stream>>>(emb, atomnum, x);

    const int elemBlocks = (ATOMS * 128 + 255) / 256; // 1024
    const int gemmTiles  = ATOMS / 16;                // 128 row-tiles
    for (int l = 0; l < 3; ++l) {
        // y = x @ in2f_W[l]   (no bias, no act)
        schnet_x2frag_k<<<elemBlocks, 256, 0, stream>>>(x, Xh);
        schnet_gemm_wmma_k<<<gemmTiles, 256, 0, stream>>>(
            Xh, I2Fh + (size_t)l * 128 * 128, nullptr, nullptr, Y, 0);
        // fused filter network + cfconv aggregation  (WMMA core)
        schnet_filter_agg_k<<<ATOMS, 256, 0, stream>>>(
            Fg, Cm, Y, neighbors,
            W1h + (size_t)l * 32 * 128, filt_b1 + (size_t)l * 128,
            W2h + (size_t)l * 128 * 128, filt_b2 + (size_t)l * 128,
            AGG);
        // t = ssp(agg @ f2out_W + f2out_b)
        schnet_x2frag_k<<<elemBlocks, 256, 0, stream>>>(AGG, Xh);
        schnet_gemm_wmma_k<<<gemmTiles, 256, 0, stream>>>(
            Xh, F2Oh + (size_t)l * 128 * 128, f2out_b + (size_t)l * 128,
            nullptr, TMP, 1);
        // x = x + t @ dense_W + dense_b
        schnet_x2frag_k<<<elemBlocks, 256, 0, stream>>>(TMP, Xh);
        schnet_gemm_wmma_k<<<gemmTiles, 256, 0, stream>>>(
            Xh, DENh + (size_t)l * 128 * 128, dense_b + (size_t)l * 128,
            x, x, 0);
    }
}
